// GATModule_7636451852431
// MI455X (gfx1250) — matrix operand order repaired and checked by reading, not compiled, not run on hardware
//
#include <hip/hip_runtime.h>
#include <hip/hip_bf16.h>
#include <stdint.h>

#define Nn    8192
#define IND   128
#define OUTD  64
#define Hh    4
#define CH    (Hh*OUTD)       // 256
#define NEG_SLOPE 0.2f
#define EPSB  1e-5f
#define MASKV -1e9f

typedef __attribute__((ext_vector_type(16))) _Float16 v16h;
typedef __attribute__((ext_vector_type(8)))  _Float16 v8h;
typedef __attribute__((ext_vector_type(8)))  float    v8f;

static __device__ __forceinline__ float lrelu(float x) { return fmaxf(x, NEG_SLOPE * x); }

// ---------------- init: zero BN stats, set ej-max accumulators to enc(-FLT_MAX)
__global__ void k_init(float* __restrict__ gsum, float* __restrict__ gsq,
                       unsigned* __restrict__ gmaxb) {
  int t = threadIdx.x;
  if (t < CH) { gsum[t] = 0.f; gsq[t] = 0.f; }
  if (t < Hh) gmaxb[t] = 0x00800000u;   // ordered-uint encoding of -FLT_MAX
}

// ---------------- projection: h = x @ W[hd], stored transposed as f16 hT[ch][n]
// W is staged into LDS PRE-SWIZZLED into the WMMA B-operand layout:
//   tile t = kb*4+ot; lane slot s = hf*16+l15; element e -> K = kb*32+16*hf+e, col = ot*16+l15
//   lds index = (t*32 + s)*16 + e   -> each B tile is ONE contiguous 32B read per lane.
__global__ __launch_bounds__(32) void k_proj(const float* __restrict__ x,
                                             const float* __restrict__ W,
                                             _Float16* __restrict__ hT) {
  __shared__ __align__(32) _Float16 wb[IND * OUTD];   // 16 KB, B-operand layout
  const int lane = threadIdx.x;
  const int nb = blockIdx.x;
  const int hd = blockIdx.y;
  const float* Wh = W + (size_t)hd * IND * OUTD;

  // stage + swizzle: i = k*64 + o (4 consecutive o share one k)
  for (int i = lane * 4; i < IND * OUTD; i += 32 * 4) {
    float4 f = *(const float4*)(Wh + i);
    int k = i >> 6;
    int o = i & 63;
    int kb  = k >> 5;
    int khf = (k >> 4) & 1;
    int e   = k & 15;
    int ot  = o >> 4;
    int l15 = o & 15;
    int base = (((kb << 2) + ot) * 32 + (khf << 4) + l15) * 16 + e;
    wb[base +  0] = (_Float16)f.x;   // l15+0
    wb[base + 16] = (_Float16)f.y;   // l15+1
    wb[base + 32] = (_Float16)f.z;   // l15+2
    wb[base + 48] = (_Float16)f.w;   // l15+3
  }
  __syncthreads();

  const int l15 = lane & 15;
  const int hf  = lane >> 4;
  const int n   = nb * 16 + l15;

  // A operand: lane row = l15; element e -> K = e + 16*hf
  v16h A[4];
  const float* xr = x + (size_t)n * IND;
  #pragma unroll
  for (int kb = 0; kb < 4; ++kb) {
    const float* p = xr + kb * 32;
    float4 p0 = *(const float4*)(p + 16 * hf);
    float4 p1 = *(const float4*)(p + 16 * hf + 4);
    float4 p2 = *(const float4*)(p + 16 * hf + 8);
    float4 p3 = *(const float4*)(p + 16 * hf + 12);
    v16h a;
    a[0]=(_Float16)p0.x; a[1]=(_Float16)p0.y; a[2]=(_Float16)p0.z; a[3]=(_Float16)p0.w;
    a[4]=(_Float16)p1.x; a[5]=(_Float16)p1.y; a[6]=(_Float16)p1.z; a[7]=(_Float16)p1.w;
    a[8]=(_Float16)p2.x; a[9]=(_Float16)p2.y; a[10]=(_Float16)p2.z; a[11]=(_Float16)p2.w;
    a[12]=(_Float16)p3.x; a[13]=(_Float16)p3.y; a[14]=(_Float16)p3.z; a[15]=(_Float16)p3.w;
    A[kb] = a;
  }

  const int slot = hf * 16 + l15;
  for (int ot = 0; ot < 4; ++ot) {
    v8f c = {};
    #pragma unroll
    for (int kb = 0; kb < 4; ++kb) {
      v16h b = *(const v16h*)(wb + (((kb << 2) + ot) * 32 + slot) * 16);
      c = __builtin_amdgcn_wmma_f32_16x16x32_f16(false, A[kb], false, b,
                                                 (short)0, c, false, false);
    }
    // C: c[r] holds row (r + 8*hf), col l15 -> 8 contiguous n's in hT[ch][n]
    v8h o;
    #pragma unroll
    for (int r = 0; r < 8; ++r) o[r] = (_Float16)c[r];
    int ch = hd * OUTD + ot * 16 + l15;
    *(v8h*)(hT + (size_t)ch * Nn + nb * 16 + 8 * hf) = o;
  }
}

// ---------------- attention scores ei/ej + per-head max(ej)
__global__ __launch_bounds__(256) void k_scores(const _Float16* __restrict__ hT,
                                                const float* __restrict__ ai,
                                                const float* __restrict__ aj,
                                                float* __restrict__ ei,
                                                float* __restrict__ ej,
                                                unsigned* __restrict__ gmaxb) {
  __shared__ float red[256];
  int hd = blockIdx.x >> 5;
  int n  = ((blockIdx.x & 31) << 8) + threadIdx.x;
  float si = 0.f, sj = 0.f;
  for (int o = 0; o < OUTD; ++o) {
    float v = (float)hT[(size_t)(hd * OUTD + o) * Nn + n];
    si += v * ai[hd * OUTD + o];
    sj += v * aj[hd * OUTD + o];
  }
  ei[(size_t)hd * Nn + n] = si;
  ej[(size_t)hd * Nn + n] = sj;
  red[threadIdx.x] = sj;
  __syncthreads();
  for (int s = 128; s > 0; s >>= 1) {
    if (threadIdx.x < s) red[threadIdx.x] = fmaxf(red[threadIdx.x], red[threadIdx.x + s]);
    __syncthreads();
  }
  if (threadIdx.x == 0) {
    unsigned u = __float_as_uint(red[0]);
    u = (u & 0x80000000u) ? ~u : (u | 0x80000000u);
    atomicMax(&gmaxb[hd], u);
  }
}

// ---------------- fused masked-softmax + aggregation (att @ h) via WMMA
#define PEL(slot, av, adjv)                                        \
  { float lg = ein + (av);                                         \
    float t  = fmaxf(lg, NEG_SLOPE * lg);                          \
    t = ((adjv) == 0) ? MASKV : t;                                 \
    float ex = __expf(t - shiftv);                                 \
    rs += ex; a[slot] = (_Float16)ex; }

__global__ __launch_bounds__(128) void k_agg(const _Float16* __restrict__ hT,
                                             const int* __restrict__ adj,
                                             const float* __restrict__ ei,
                                             const float* __restrict__ ej,
                                             const unsigned* __restrict__ gmaxb,
                                             float* __restrict__ out) {
  __shared__ float rsum_s[Hh][16];
  const int tid  = threadIdx.x;
  const int hd   = tid >> 5;       // one head per wave
  const int lane = tid & 31;
  const int l15  = lane & 15;
  const int hf   = lane >> 4;
  const int nb   = blockIdx.x;
  const int n    = nb * 16 + l15;  // A-operand row handled by this lane

  unsigned mu = gmaxb[hd];
  float gm = (mu & 0x80000000u) ? __uint_as_float(mu & 0x7fffffffu)
                                : __uint_as_float(~mu);
  float ein    = ei[(size_t)hd * Nn + n];
  float shiftv = lrelu(ein + gm);   // upper bound of row logits (lrelu monotone)

  const int*   arow = adj + (size_t)n * Nn;
  const float* ejh  = ej + (size_t)hd * Nn;
  const _Float16* hTh = hT + (size_t)hd * OUTD * Nn;

  v8f acc[4]; acc[0] = (v8f){}; acc[1] = (v8f){}; acc[2] = (v8f){}; acc[3] = (v8f){};
  float rs = 0.f;

  for (int m = 0; m < Nn; m += 32) {
    // pull the adjacency stream ahead (global_prefetch_b8); speculative => safe
    if (m + 256 < Nn) __builtin_prefetch(arow + m + 256, 0, 1);

    // adjacency + ej for this lane's A-layout K indices: run [m+16h, +16)
    int4   q0 = *(const int4*)(arow + m + 16 * hf);
    int4   q1 = *(const int4*)(arow + m + 16 * hf + 4);
    int4   q2 = *(const int4*)(arow + m + 16 * hf + 8);
    int4   q3 = *(const int4*)(arow + m + 16 * hf + 12);
    float4 e0 = *(const float4*)(ejh + m + 16 * hf);
    float4 e1 = *(const float4*)(ejh + m + 16 * hf + 4);
    float4 e2 = *(const float4*)(ejh + m + 16 * hf + 8);
    float4 e3 = *(const float4*)(ejh + m + 16 * hf + 12);

    v16h a;
    PEL(0,  e0.x, q0.x) PEL(1,  e0.y, q0.y) PEL(2,  e0.z, q0.z) PEL(3,  e0.w, q0.w)
    PEL(4,  e1.x, q1.x) PEL(5,  e1.y, q1.y) PEL(6,  e1.z, q1.z) PEL(7,  e1.w, q1.w)
    PEL(8,  e2.x, q2.x) PEL(9,  e2.y, q2.y) PEL(10, e2.z, q2.z) PEL(11, e2.w, q2.w)
    PEL(12, e3.x, q3.x) PEL(13, e3.y, q3.y) PEL(14, e3.z, q3.z) PEL(15, e3.w, q3.w)

    // B operand from hT: lane col = l15, K = e + 16*hf -> 16 contiguous f16 (32B)
    #pragma unroll
    for (int ot = 0; ot < 4; ++ot) {
      v16h b = *(const v16h*)(hTh + (size_t)(ot * 16 + l15) * Nn + m + 16 * hf);
      acc[ot] = __builtin_amdgcn_wmma_f32_16x16x32_f16(false, a, false, b,
                                                       (short)0, acc[ot], false, false);
    }
  }

  float tot = rs + __shfl_xor(rs, 16);   // combine the two K-half partial row sums
  if (lane < 16) rsum_s[hd][l15] = tot;
  __syncthreads();

  #pragma unroll
  for (int ot = 0; ot < 4; ++ot) {
    #pragma unroll
    for (int r = 0; r < 8; ++r) {
      int row = 8 * hf + r;
      float v = acc[ot][r] / rsum_s[hd][row];
      out[(size_t)(nb * 16 + row) * CH + hd * OUTD + ot * 16 + l15] = v;
    }
  }
}

// ---------------- BN statistics (sum / sumsq per channel)
__global__ __launch_bounds__(256) void k_stats(const float* __restrict__ out,
                                               float* __restrict__ gsum,
                                               float* __restrict__ gsq) {
  int c  = threadIdx.x;
  int r0 = blockIdx.x * 128;
  float s = 0.f, s2 = 0.f;
  for (int r = 0; r < 128; ++r) {
    float v = out[(size_t)(r0 + r) * CH + c];
    s += v; s2 += v * v;
  }
  atomicAdd(&gsum[c], s);
  atomicAdd(&gsq[c], s2);
}

// ---------------- BN apply + ReLU, in place on d_out
__global__ __launch_bounds__(256) void k_bn(float* __restrict__ out,
                                            const float* __restrict__ gsum,
                                            const float* __restrict__ gsq,
                                            const float* __restrict__ gamma,
                                            const float* __restrict__ beta) {
  size_t i = (size_t)blockIdx.x * 256 + threadIdx.x;
  int c = (int)(i & (CH - 1));
  const float inv = 1.0f / (float)Nn;
  float mean = gsum[c] * inv;
  float var  = gsq[c] * inv - mean * mean;
  float v = out[i];
  v = (v - mean) * rsqrtf(var + EPSB) * gamma[c] + beta[c];
  out[i] = fmaxf(v, 0.f);
}

extern "C" void kernel_launch(void* const* d_in, const int* in_sizes, int n_in,
                              void* d_out, int out_size, void* d_ws, size_t ws_size,
                              hipStream_t stream) {
  const float* x     = (const float*)d_in[0];
  const int*   adj   = (const int*)d_in[1];
  const float* W     = (const float*)d_in[2];
  const float* a_i   = (const float*)d_in[3];
  const float* a_j   = (const float*)d_in[4];
  const float* gamma = (const float*)d_in[5];
  const float* beta  = (const float*)d_in[6];
  float* out = (float*)d_out;

  char* ws = (char*)d_ws;
  _Float16* hT = (_Float16*)ws;
  size_t off = (size_t)CH * Nn * sizeof(_Float16);        // 4 MB
  float* ei   = (float*)(ws + off); off += (size_t)Hh * Nn * 4;
  float* ej   = (float*)(ws + off); off += (size_t)Hh * Nn * 4;
  float* gsum = (float*)(ws + off); off += CH * 4;
  float* gsq  = (float*)(ws + off); off += CH * 4;
  unsigned* gmaxb = (unsigned*)(ws + off);

  hipLaunchKernelGGL(k_init,   dim3(1),            dim3(256), 0, stream, gsum, gsq, gmaxb);
  hipLaunchKernelGGL(k_proj,   dim3(Nn / 16, Hh),  dim3(32),  0, stream, x, W, hT);
  hipLaunchKernelGGL(k_scores, dim3(Hh * 32),      dim3(256), 0, stream, hT, a_i, a_j, ei, ej, gmaxb);
  hipLaunchKernelGGL(k_agg,    dim3(Nn / 16),      dim3(128), 0, stream, hT, adj, ei, ej, gmaxb, out);
  hipLaunchKernelGGL(k_stats,  dim3(Nn / 128),     dim3(256), 0, stream, out, gsum, gsq);
  hipLaunchKernelGGL(k_bn,     dim3(Nn * CH / 256), dim3(256), 0, stream, out, gsum, gsq, gamma, beta);
}
